// MultiHeadAttention_13237089206671
// MI455X (gfx1250) — matrix operand order repaired and checked by reading, not compiled, not run on hardware
//
#include <hip/hip_runtime.h>
#include <hip/hip_bf16.h>

typedef __attribute__((ext_vector_type(16))) _Float16 v16h;
typedef __attribute__((ext_vector_type(8)))  _Float16 v8h;
typedef __attribute__((ext_vector_type(8)))  float    v8f;

typedef unsigned int uint32x4 __attribute__((ext_vector_type(4)));
typedef int          int32x8  __attribute__((ext_vector_type(8)));
typedef int          int32x4  __attribute__((ext_vector_type(4)));

#define S_LEN   2048
#define D_MODEL 512
#define NHEADS  8
#define DK      64
#define MROWS   (4 * S_LEN)   // B*S = 8192

#if __has_builtin(__builtin_amdgcn_tensor_load_to_lds) && \
    __has_builtin(__builtin_amdgcn_s_wait_tensorcnt)
#define USE_TDM 1
#else
#define USE_TDM 0
#endif

// ---------------------------------------------------------------------------
// Fragment helpers (CDNA5 WMMA VGPR layouts, CDNA5 ISA §7.12.2)
// ---------------------------------------------------------------------------

// A-matrix 16x32 (MxK) fragment. base is LDS tile [rows][ldh] (halfs).
// lanes 0-15 : row m, elems 0-7 = K0..7,  elems 8-15 = K16..23
// lanes 16-31: row m, elems 0-7 = K8..15, elems 8-15 = K24..31
__device__ __forceinline__ v16h load_frag_a(const _Float16* base, int row0,
                                            int ldh, int kk0, int lane) {
  int m    = row0 + (lane & 15);
  int koff = (lane >> 4) * 8;
  const _Float16* p = base + m * ldh + kk0 + koff;
  v8h lo = *(const v8h*)(p);
  v8h hi = *(const v8h*)(p + 16);
  return __builtin_shufflevector(lo, hi, 0, 1, 2, 3, 4, 5, 6, 7,
                                          8, 9, 10, 11, 12, 13, 14, 15);
}

// B-matrix 32x16 (KxN) fragment from a TRANSPOSED LDS tile: base[n*ldh + k].
// lanes 0-15 : col n, elems 0-7 = K0..7, 8-15 = K16..23 ; lanes 16-31: K8..15, K24..31
__device__ __forceinline__ v16h load_frag_b(const _Float16* base, int col0,
                                            int ldh, int kk0, int lane) {
  int n  = col0 + (lane & 15);
  int kb = kk0 + (lane >> 4) * 8;
  const _Float16* p = base + n * ldh + kb;
  v8h lo = *(const v8h*)(p);
  v8h hi = *(const v8h*)(p + 16);
  return __builtin_shufflevector(lo, hi, 0, 1, 2, 3, 4, 5, 6, 7,
                                          8, 9, 10, 11, 12, 13, 14, 15);
}

__device__ __forceinline__ v8f wmma_f16(v16h a, v16h b, v8f c) {
  return __builtin_amdgcn_wmma_f32_16x16x32_f16(false, a, false, b,
                                                (short)0, c, false, false);
}

#if USE_TDM
// ---------------------------------------------------------------------------
// Tensor Data Mover: 2D tile load Global->LDS (CDNA5 ISA)
//   data_size_log2: 0=1B,1=2B,2=4B    pad codes per D# group1 encoding:
//   pad_interval: pad after 2^(code+1) DWORDs ; pad_amount: (code+1) DWORDs
// This toolchain exposes the 6-arg builtin form:
//   (uint32x4 g0, int32x8 g1, int32x4 g2, int32x4 g3, int32x8, i32 cpol)
// ---------------------------------------------------------------------------
__device__ __forceinline__ void tdm_load_2d(const void* gptr, unsigned lds_off,
                                            unsigned data_size_log2,
                                            unsigned tile_d0, unsigned tile_d1,
                                            unsigned tensor_d0, unsigned tensor_d1,
                                            unsigned long long stride_elems,
                                            bool pad_en,
                                            unsigned pad_interval_code,
                                            unsigned pad_amount_code) {
  unsigned long long ga = (unsigned long long)gptr;
  uint32x4 g0;
  g0.x = 0x1u;                                        // count=1, user mode
  g0.y = lds_off;                                     // lds_addr [63:32]
  g0.z = (unsigned)(ga & 0xFFFFFFFFu);                // global_addr lo
  g0.w = (unsigned)((ga >> 32) & 0x01FFFFFFu)         // global_addr[56:32]
         | 0x80000000u;                               // type=2 ("image")
  int32x8 g1;
  g1[0] = (int)((data_size_log2 << 16) |
                (pad_en ? (1u << 20) : 0u) |
                (pad_interval_code << 22) |
                (pad_amount_code << 25));
  g1[1] = (int)((tensor_d0 & 0xFFFFu) << 16);                     // dim0 lo16
  g1[2] = (int)(((tensor_d0 >> 16) & 0xFFFFu) |
                ((tensor_d1 & 0xFFFFu) << 16));                   // dim0 hi/dim1 lo
  g1[3] = (int)(((tensor_d1 >> 16) & 0xFFFFu) |
                ((tile_d0 & 0xFFFFu) << 16));                     // dim1 hi/tile0
  g1[4] = (int)(tile_d1 & 0xFFFFu);                               // tile1, tile2=0
  g1[5] = (int)(unsigned)(stride_elems & 0xFFFFFFFFu);            // dim0_stride lo
  g1[6] = (int)(unsigned)((stride_elems >> 32) & 0xFFFFu);        // stride hi16
  g1[7] = 0;
  int32x4 z4 = {0, 0, 0, 0};
  int32x8 z8 = {0, 0, 0, 0, 0, 0, 0, 0};
  __builtin_amdgcn_tensor_load_to_lds(g0, g1, z4, z4, z8, 0);
}

__device__ __forceinline__ unsigned lds_offset_of(const void* p) {
  return (unsigned)(unsigned long long)p;   // low 32 bits of flat LDS address
}
#endif

// ---------------------------------------------------------------------------
// GEMM: Y = X @ W^T + bias.  M=8192, N=K=512.
// Block = 256 threads (8 waves, 4x2), tile 128x64, K-step 32.
// ---------------------------------------------------------------------------
template <bool IN_F16, bool OUT_HEADS>
__global__ __launch_bounds__(256)
void mha_gemm512_kernel(const void* __restrict__ Xv,
                        const float* __restrict__ W,
                        const float* __restrict__ bias,
                        void* __restrict__ Yv) {
  __shared__ __align__(16) _Float16 Asb[128 * 40];  // 128x32, pad->40 (80B rows)
  __shared__ __align__(16) _Float16 Bsb[64 * 40];   // W tile transposed: [n][k]

  const int bm = blockIdx.x;
  const int bn = blockIdx.y;
  const int t  = threadIdx.x;
  const int w  = t >> 5;
  const int lane = t & 31;
  const int wm = w & 3;
  const int wn = w >> 2;

  const float*    Xf = (const float*)Xv;
  const _Float16* Xh = (const _Float16*)Xv;

  v8f acc[2][2] = {};

  for (int k0 = 0; k0 < D_MODEL; k0 += 32) {
#pragma unroll
    for (int it = 0; it < 16; ++it) {
      int e  = t + it * 256;
      int i  = e >> 5;
      int kk = e & 31;
      size_t g = (size_t)(bm * 128 + i) * D_MODEL + k0 + kk;
      float x = IN_F16 ? (float)Xh[g] : Xf[g];
      Asb[i * 40 + kk] = (_Float16)x;
    }
#pragma unroll
    for (int it = 0; it < 8; ++it) {
      int e  = t + it * 256;
      int nn = e >> 5;
      int kk = e & 31;
      Bsb[nn * 40 + kk] = (_Float16)W[(size_t)(bn * 64 + nn) * D_MODEL + k0 + kk];
    }
    __syncthreads();

    v16h af[2], bf[2];
    af[0] = load_frag_a(Asb, wm * 32,      40, 0, lane);
    af[1] = load_frag_a(Asb, wm * 32 + 16, 40, 0, lane);
    bf[0] = load_frag_b(Bsb, wn * 32,      40, 0, lane);
    bf[1] = load_frag_b(Bsb, wn * 32 + 16, 40, 0, lane);
#pragma unroll
    for (int mi = 0; mi < 2; ++mi)
#pragma unroll
      for (int ni = 0; ni < 2; ++ni)
        acc[mi][ni] = wmma_f16(af[mi], bf[ni], acc[mi][ni]);
    __syncthreads();
  }

  const int colb  = lane & 15;
  const int rhalf = (lane >> 4) * 8;
#pragma unroll
  for (int mi = 0; mi < 2; ++mi) {
#pragma unroll
    for (int ni = 0; ni < 2; ++ni) {
#pragma unroll
      for (int r = 0; r < 8; ++r) {
        int grow = bm * 128 + wm * 32 + mi * 16 + r + rhalf;
        int gcol = bn * 64 + wn * 32 + ni * 16 + colb;
        float val = acc[mi][ni][r] + bias[gcol];
        if (OUT_HEADS) {
          int b = grow >> 11, s = grow & (S_LEN - 1);
          int h = gcol >> 6, d = gcol & (DK - 1);
          ((_Float16*)Yv)[((size_t)(b * NHEADS + h) * S_LEN + s) * DK + d] =
              (_Float16)val;
        } else {
          ((float*)Yv)[(size_t)grow * D_MODEL + gcol] = val;
        }
      }
    }
  }
}

// ---------------------------------------------------------------------------
// Flash attention per (b,h): 64 query rows per block, KV chunks of 64.
// Q/K/mask tiles streamed by the Tensor Data Mover (wave 0 issues the D#,
// hardware applies the LDS row padding); V staged manually (needs transpose).
// ---------------------------------------------------------------------------
__global__ __launch_bounds__(256)
void mha_flash_attn_kernel(const _Float16* __restrict__ Qh,
                           const _Float16* __restrict__ Kh,
                           const _Float16* __restrict__ Vh,
                           const int* __restrict__ mask,
                           _Float16* __restrict__ Oout) {
  __shared__ __align__(16) _Float16 Qs[64 * 72];   // [q][d]
  __shared__ __align__(16) _Float16 Ks[64 * 72];   // [kv][d]   (B-frag of QK^T)
  __shared__ __align__(16) _Float16 Vt[64 * 72];   // [d][kv]   (B-frag of P@V)
  __shared__ __align__(16) _Float16 Ps[64 * 72];   // [q][kv]   probs
  __shared__ __align__(16) int      Ms[64 * 65];   // mask tile (TDM-padded rows)
  __shared__ float Ss[64 * 65];                    // raw scores
  __shared__ float m_run[64], l_run[64], alpha_s[64];

  const int q0 = blockIdx.x * 64;
  const int bh = blockIdx.y;
  const int b = bh >> 3, h = bh & 7;

  const _Float16* Qb = Qh + (size_t)bh * S_LEN * DK;
  const _Float16* Kb = Kh + (size_t)bh * S_LEN * DK;
  const _Float16* Vb = Vh + (size_t)bh * S_LEN * DK;

  const int t = threadIdx.x;
  const int w = t >> 5;
  const int lane = t & 31;
  const int wm = w & 3;
  const int wn = w >> 2;
  const int rhalf = (lane >> 4) * 8;

  // ---- Stage Q tile (once) ----
#if USE_TDM
  if (w == 0) {
    // 64x64 f16 tile; 32-DWORD rows (code 4) + 4-DWORD pad (code 3) -> 72-half rows
    tdm_load_2d(Qb + (size_t)q0 * DK, lds_offset_of(Qs),
                /*dsz*/1, /*tile*/DK, 64, /*tensor*/DK, S_LEN,
                /*stride*/DK, /*pad*/true, 4, 3);
    __builtin_amdgcn_s_wait_tensorcnt(0);
  }
#else
#pragma unroll
  for (int it = 0; it < 16; ++it) {
    int e = t + it * 256;
    int i = e >> 6, d = e & 63;
    Qs[i * 72 + d] = Qb[(size_t)(q0 + i) * DK + d];
  }
#endif
  if (t < 64) { m_run[t] = -3.0e38f; l_run[t] = 0.f; }
  __syncthreads();

  // Q fragments are reused by every KV chunk: keep them in registers.
  v16h aq0 = load_frag_a(Qs, wm * 16, 72, 0,  lane);
  v16h aq1 = load_frag_a(Qs, wm * 16, 72, 32, lane);

  v8f oacc[2] = {};

  for (int kv0 = 0; kv0 < S_LEN; kv0 += 64) {
    __syncthreads();   // previous chunk done with Ks/Vt/Ps/Ss/Ms

    // ---- Stage K chunk + mask tile (TDM) and V chunk transposed (manual) ----
#if USE_TDM
    if (w == 0) {
      tdm_load_2d(Kb + (size_t)kv0 * DK, lds_offset_of(Ks),
                  /*dsz*/1, /*tile*/DK, 64, /*tensor*/DK, S_LEN,
                  /*stride*/DK, /*pad*/true, 4, 3);
      // 64x64 int32 tile; 64-DWORD rows (code 5) + 1-DWORD pad (code 0) -> 65 ints
      tdm_load_2d(mask + (size_t)q0 * S_LEN + kv0, lds_offset_of(Ms),
                  /*dsz*/2, /*tile*/64, 64, /*tensor*/S_LEN, S_LEN,
                  /*stride*/S_LEN, /*pad*/true, 5, 0);
    }
#else
#pragma unroll
    for (int it = 0; it < 16; ++it) {
      int e = t + it * 256;
      int i = e >> 6, d = e & 63;
      Ks[i * 72 + d] = Kb[(size_t)(kv0 + i) * DK + d];
      Ms[i * 65 + d] = mask[(size_t)(q0 + i) * S_LEN + kv0 + d];
    }
#endif
#pragma unroll
    for (int it = 0; it < 16; ++it) {
      int e = t + it * 256;
      int i = e >> 6, d = e & 63;
      Vt[d * 72 + i] = Vb[(size_t)(kv0 + i) * DK + d];
    }
    if (kv0 + 64 < S_LEN && t < 64) {
      __builtin_prefetch(Vb + (size_t)(kv0 + 64 + t) * DK, 0, 1);
    }
#if USE_TDM
    if (w == 0) __builtin_amdgcn_s_wait_tensorcnt(0);
#endif
    __syncthreads();

    // ---- Scores: S = (Q @ K^T) * 1/sqrt(dk).  Wave covers 16q x 32kv ----
#pragma unroll
    for (int ni = 0; ni < 2; ++ni) {
      v8f sac = {};
      v16h bk0 = load_frag_b(Ks, wn * 32 + ni * 16, 72, 0,  lane);
      v16h bk1 = load_frag_b(Ks, wn * 32 + ni * 16, 72, 32, lane);
      sac = wmma_f16(aq0, bk0, sac);
      sac = wmma_f16(aq1, bk1, sac);
#pragma unroll
      for (int r = 0; r < 8; ++r) {
        int row = wm * 16 + r + rhalf;
        int col = wn * 32 + ni * 16 + (lane & 15);
        Ss[row * 65 + col] = sac[r] * 0.125f;   // 1/sqrt(64)
      }
    }
    __syncthreads();

    // ---- Online softmax row pass (one thread per query row) ----
    if (t < 64) {
      int q = t;
      float mold = m_run[q];
      float mc = -3.0e38f;
#pragma unroll 4
      for (int c = 0; c < 64; ++c) {
        float s = Ss[q * 65 + c];
        s = (Ms[q * 65 + c] == 0) ? -1.0e9f : s;
        mc = fmaxf(mc, s);
      }
      float mnew = fmaxf(mold, mc);
      float al = __expf(mold - mnew);
      float sum = 0.f;
#pragma unroll 4
      for (int c = 0; c < 64; ++c) {
        float s = Ss[q * 65 + c];
        s = (Ms[q * 65 + c] == 0) ? -1.0e9f : s;
        float p = __expf(s - mnew);
        sum += p;
        Ps[q * 72 + c] = (_Float16)p;
      }
      l_run[q] = l_run[q] * al + sum;
      m_run[q] = mnew;
      alpha_s[q] = al;
    }
    __syncthreads();

    // ---- Rescale accumulator, then O += P @ V.  Wave covers 16q x 32dk ----
#pragma unroll
    for (int ni = 0; ni < 2; ++ni)
#pragma unroll
      for (int r = 0; r < 8; ++r)
        oacc[ni][r] *= alpha_s[wm * 16 + r + rhalf];

    v16h ap0 = load_frag_a(Ps, wm * 16, 72, 0,  lane);
    v16h ap1 = load_frag_a(Ps, wm * 16, 72, 32, lane);
#pragma unroll
    for (int ni = 0; ni < 2; ++ni) {
      v16h bv0 = load_frag_b(Vt, wn * 32 + ni * 16, 72, 0,  lane);
      v16h bv1 = load_frag_b(Vt, wn * 32 + ni * 16, 72, 32, lane);
      oacc[ni] = wmma_f16(ap0, bv0, oacc[ni]);
      oacc[ni] = wmma_f16(ap1, bv1, oacc[ni]);
    }
  }

  // ---- Epilogue: divide by l, store f16 into [B,S,D] with col = h*64+d ----
#pragma unroll
  for (int ni = 0; ni < 2; ++ni) {
#pragma unroll
    for (int r = 0; r < 8; ++r) {
      int row = wm * 16 + r + rhalf;
      int d   = wn * 32 + ni * 16 + (lane & 15);
      float l = fmaxf(l_run[row], 1e-30f);
      float val = oacc[ni][r] / l;
      Oout[((size_t)(b * S_LEN + q0 + row)) * D_MODEL + h * DK + d] =
          (_Float16)val;
    }
  }
}

// ---------------------------------------------------------------------------
extern "C" void kernel_launch(void* const* d_in, const int* in_sizes, int n_in,
                              void* d_out, int out_size, void* d_ws,
                              size_t ws_size, hipStream_t stream) {
  const float* q    = (const float*)d_in[0];
  const float* k    = (const float*)d_in[1];
  const float* v    = (const float*)d_in[2];
  const int*   mask = (const int*)d_in[3];
  const float* Wq = (const float*)d_in[4];
  const float* bq = (const float*)d_in[5];
  const float* Wk = (const float*)d_in[6];
  const float* bk = (const float*)d_in[7];
  const float* Wv = (const float*)d_in[8];
  const float* bv = (const float*)d_in[9];
  const float* Wo = (const float*)d_in[10];
  const float* bo = (const float*)d_in[11];

  char* ws = (char*)d_ws;
  const size_t HEAD_BYTES = (size_t)4 * NHEADS * S_LEN * DK * sizeof(_Float16); // 8 MiB
  _Float16* Qh = (_Float16*)(ws);
  _Float16* Kh = (_Float16*)(ws + HEAD_BYTES);
  _Float16* Vh = (_Float16*)(ws + 2 * HEAD_BYTES);
  _Float16* Ao = (_Float16*)(ws + 3 * HEAD_BYTES);   // [8192,512] f16

  dim3 gemm_grid(MROWS / 128, D_MODEL / 64);   // (64, 8)
  dim3 block(256);

  mha_gemm512_kernel<false, true><<<gemm_grid, block, 0, stream>>>(
      (const void*)q, Wq, bq, (void*)Qh);
  mha_gemm512_kernel<false, true><<<gemm_grid, block, 0, stream>>>(
      (const void*)k, Wk, bk, (void*)Kh);
  mha_gemm512_kernel<false, true><<<gemm_grid, block, 0, stream>>>(
      (const void*)v, Wv, bv, (void*)Vh);

  dim3 attn_grid(S_LEN / 64, 4 * NHEADS);      // (32, 32)
  mha_flash_attn_kernel<<<attn_grid, block, 0, stream>>>(Qh, Kh, Vh, mask, Ao);

  mha_gemm512_kernel<true, false><<<gemm_grid, block, 0, stream>>>(
      (const void*)Ao, Wo, bo, d_out);
}
